// MixtureOfExperts_50096498540668
// MI455X (gfx1250) — matrix operand order repaired and checked by reading, not compiled, not run on hardware
//
#include <hip/hip_runtime.h>
#include <hip/hip_bf16.h>
#include <math.h>

#define N_TOKENS 32768
#define D_MODEL  1024
#define D_HIDDEN 4096
#define N_EXPERTS 8

#if defined(__gfx1250__) && __has_builtin(__builtin_amdgcn_global_load_async_to_lds_b128)
#define HAVE_ASYNC_LDS 1
#endif

// ---------------------------------------------------------------------------
// WMMA types (CDNA5 gfx1250, wave32): D(16x16 f32) = A(16x32 bf16) x B(32x16
// bf16) + C.  A: one row per lane (lanes 0-15: K 0-7 & 16-23, lanes 16-31:
// K 8-15 & 24-31).  B: one column per lane (lanes 0-15: K 0-7 & 16-23, lanes 16-31:
// K 8-15 & 24-31; 2 consecutive K per VGPR).  C/D: N = lane%16, M = v (+8 if lane>=16).
// ---------------------------------------------------------------------------
typedef __attribute__((ext_vector_type(16))) __bf16 v16bf;
typedef __attribute__((ext_vector_type(8)))  float  v8f;
typedef __attribute__((ext_vector_type(4)))  int    v4i;

union AFrag { v16bf v; uint4 q[2]; };
union BFrag { v16bf v; uint4 q[2]; };

static __device__ __forceinline__ unsigned pack_bf16(float a, float b) {
  unsigned ua = __float_as_uint(a), ub = __float_as_uint(b);
  ua = (ua + 0x7FFFu + ((ua >> 16) & 1u)) >> 16;   // RNE f32 -> bf16
  ub = (ub + 0x7FFFu + ((ub >> 16) & 1u)) >> 16;
  return ua | (ub << 16);
}

static __device__ __forceinline__ float gelu_exact(float v) {
  return 0.5f * v * (1.0f + erff(v * 0.70710678118654752f));
}

// ---------------------------------------------------------------------------
// Kernel 1: zero output (scatter phase uses atomic f32 adds) + expert counters
// ---------------------------------------------------------------------------
__global__ void moe_zero_kernel(float4* __restrict__ out4, int n4,
                                int* __restrict__ cnt) {
  int gid = blockIdx.x * blockDim.x + threadIdx.x;
  int stride = gridDim.x * blockDim.x;
  float4 z = make_float4(0.f, 0.f, 0.f, 0.f);
  for (int i = gid; i < n4; i += stride) out4[i] = z;
  if (gid < N_EXPERTS) cnt[gid] = 0;
}

// ---------------------------------------------------------------------------
// Kernel 2: x (fp32) -> bf16 copy in workspace (2 elems per uint)
// ---------------------------------------------------------------------------
__global__ void moe_cvtx_kernel(const float* __restrict__ x,
                                unsigned* __restrict__ xb, int n2) {
  int gid = blockIdx.x * blockDim.x + threadIdx.x;
  int stride = gridDim.x * blockDim.x;
  for (int i = gid; i < n2; i += stride)
    xb[i] = pack_bf16(x[2 * i], x[2 * i + 1]);
}

// ---------------------------------------------------------------------------
// Kernel 3: swizzle weights [E][K][N] fp32 -> WMMA-B-fragment-native bf16.
// Fragment tile (kt, nt) covers K rows [kt*32,+32), N cols [nt*16,+16) and is
// stored as a contiguous 1KB block at u32 offset ((nt*KT + kt)*32 + lane)*8,
// where lane = half*16 + (n%16) and the 8 u32 hold K-pairs (K = (v/4)*16+half*8+2*(v%4)).
// A consuming wave reads a whole fragment as one coalesced 1KB burst
// (2 x b128 per lane) with zero repack VALU.
// ---------------------------------------------------------------------------
__global__ void moe_cvtw_kernel(const float* __restrict__ W,
                                unsigned* __restrict__ Wf, int K, int N) {
  const int KT = K >> 5;           // k tiles of 32
  const int kn2 = (K * N) >> 1;    // u32s per expert
  const int total = N_EXPERTS * kn2;
  int gid = blockIdx.x * blockDim.x + threadIdx.x;
  int stride = gridDim.x * blockDim.x;
  for (int f = gid; f < total; f += stride) {
    int e = f / kn2;
    int rem = f - e * kn2;
    int v = rem & 7;
    int lane = (rem >> 3) & 31;
    int tmp = rem >> 8;
    int kt = tmp % KT;
    int nt = tmp / KT;
    int k = kt * 32 + (v >> 2) * 16 + (lane >> 4) * 8 + 2 * (v & 3);
    int n = nt * 16 + (lane & 15);
    const float* src = W + ((size_t)e * K + k) * N + n;
    Wf[f] = pack_bf16(src[0], src[N]);
  }
}

// ---------------------------------------------------------------------------
// Kernel 4: router — one wave32 per token. logits = x@Wr + br, exact top-2
// softmax renorm, append (token, weight) to per-expert lists via atomics.
// ---------------------------------------------------------------------------
__global__ __launch_bounds__(256)
void moe_router_kernel(const float* __restrict__ x, const float* __restrict__ Wr,
                       const float* __restrict__ br, int* __restrict__ cnt,
                       int* __restrict__ idx, float* __restrict__ wgt) {
  const int t = blockIdx.x * 8 + (threadIdx.x >> 5);
  const int lane = threadIdx.x & 31;

  float acc[N_EXPERTS];
#pragma unroll
  for (int e = 0; e < N_EXPERTS; ++e) acc[e] = 0.0f;

  const float* xr = x + (size_t)t * D_MODEL;
  for (int k = lane; k < D_MODEL; k += 32) {
    float xv = xr[k];
    const float4* wr = (const float4*)(Wr + (size_t)k * N_EXPERTS);
    float4 wa = wr[0], wb = wr[1];
    acc[0] += xv * wa.x; acc[1] += xv * wa.y;
    acc[2] += xv * wa.z; acc[3] += xv * wa.w;
    acc[4] += xv * wb.x; acc[5] += xv * wb.y;
    acc[6] += xv * wb.z; acc[7] += xv * wb.w;
  }
#pragma unroll
  for (int off = 16; off >= 1; off >>= 1)
#pragma unroll
    for (int e = 0; e < N_EXPERTS; ++e)
      acc[e] += __shfl_down(acc[e], off, 32);

  if (lane == 0) {
    float l[N_EXPERTS];
#pragma unroll
    for (int e = 0; e < N_EXPERTS; ++e) l[e] = acc[e] + br[e];
    int e0 = 0;
#pragma unroll
    for (int e = 1; e < N_EXPERTS; ++e) if (l[e] > l[e0]) e0 = e;
    int e1 = -1;
#pragma unroll
    for (int e = 0; e < N_EXPERTS; ++e)
      if (e != e0 && (e1 < 0 || l[e] > l[e1])) e1 = e;
    // renormalized top-2 softmax == softmax over {l0, l1}
    float w0 = 1.0f / (1.0f + expf(l[e1] - l[e0]));
    float w1 = 1.0f - w0;
    int p0 = atomicAdd(&cnt[e0], 1);
    idx[e0 * N_TOKENS + p0] = t; wgt[e0 * N_TOKENS + p0] = w0;
    int p1 = atomicAdd(&cnt[e1], 1);
    idx[e1 * N_TOKENS + p1] = t; wgt[e1 * N_TOKENS + p1] = w1;
  }
}

// ---------------------------------------------------------------------------
// Kernel 5: fused per-expert MLP over a 64-token gathered tile.
//   8 waves / block. Gathered x tile (64x1024 bf16 = 128KB) staged in LDS
//   via gfx1250 async-to-LDS loads (ASYNCcnt) when available; A fragments
//   stream from LDS via ds_load_b128. Hidden processed in chunks of 64:
//   phase1 h=gelu(x@W1c+b1): each wave owns a 32-row pair x one 16-col tile
//   -> every B fragment feeds 2 WMMAs. Phase2 out += h@W2c: each wave owns
//   all 64 rows x 128 cols -> every B fragment feeds 4 WMMAs (oacc[4][8] =
//   256 VGPRs). B fragments stream from pre-swizzled bf16 weights as
//   coalesced 1KB bursts. Epilogue: *wt, +b2, atomic scatter-add.
// ---------------------------------------------------------------------------
#define SM_XA    0
#define SM_HBUF  (64 * 1024 * 2)                   // 131072
#define SM_TOK   (SM_HBUF + 64 * 64 * 2)           // +8192
#define SM_WT    (SM_TOK + 64 * 4)                 // +256
#define SM_TOTAL (SM_WT + 64 * 4)                  // 139776 bytes

__global__ __launch_bounds__(256, 1)
void moe_expert_kernel(const unsigned short* __restrict__ xb_raw,
                       const uint4* __restrict__ W1f, const float* __restrict__ b1,
                       const uint4* __restrict__ W2f, const float* __restrict__ b2,
                       const int* __restrict__ cnt, const int* __restrict__ idx,
                       const float* __restrict__ wgt, float* __restrict__ out) {
  const int e = blockIdx.y;
  const int base = blockIdx.x * 64;
  const int count = cnt[e];
  if (base >= count) return;
  const int rows = (count - base < 64) ? (count - base) : 64;

  extern __shared__ __align__(16) char smem[];
  __bf16 (*XA)[D_MODEL] = (__bf16(*)[D_MODEL])(smem + SM_XA);   // [64][1024]
  __bf16 (*Hbuf)[64]    = (__bf16(*)[64])(smem + SM_HBUF);      // [64][64]
  int*   tok_s          = (int*)(smem + SM_TOK);
  float* wt_s           = (float*)(smem + SM_WT);

  const int tid  = threadIdx.x;
  const int lane = tid & 31;
  const int w    = tid >> 5;     // wave id 0..7
  const int r    = lane & 15;
  const int half = lane >> 4;

  if (tid < 64) {
    int gi = e * N_TOKENS + base + tid;
    tok_s[tid] = (tid < rows) ? idx[gi] : idx[e * N_TOKENS + base];
    wt_s[tid]  = (tid < rows) ? wgt[gi] : 0.0f;   // padded rows add 0
  }
  __syncthreads();

  // Cooperative gather of 64 token rows (2KB each) into LDS: 4 threads/row,
  // 512B contiguous per thread.
  {
    const int row = tid >> 2, seg = tid & 3;
    const __bf16* xb = (const __bf16*)xb_raw;
    const uint4* src = (const uint4*)(xb + (size_t)tok_s[row] * D_MODEL) + seg * 32;
    uint4* dst = (uint4*)(&XA[row][0]) + seg * 32;
#if defined(HAVE_ASYNC_LDS)
    // gfx1250 async DMA: memory -> LDS without VGPR staging (ASYNCcnt).
#pragma unroll 8
    for (int i = 0; i < 32; ++i)
      __builtin_amdgcn_global_load_async_to_lds_b128(
          (__attribute__((address_space(1))) v4i*)(src + i),
          (__attribute__((address_space(3))) v4i*)(dst + i),
          0, 0);
#if __has_builtin(__builtin_amdgcn_s_wait_asynccnt)
    __builtin_amdgcn_s_wait_asynccnt(0);
#else
    asm volatile("s_wait_asynccnt 0x0" ::: "memory");
#endif
#else
#pragma unroll 8
    for (int i = 0; i < 32; ++i) dst[i] = src[i];
#endif
  }
  __syncthreads();

  // Per-expert fragment bases (uint4 units; fragment = 64 uint4 = 1KB).
  const uint4* W1fe = W1f + (size_t)e * (D_MODEL / 32) * (D_HIDDEN / 16) * 64;
  const uint4* W2fe = W2f + (size_t)e * (D_HIDDEN / 32) * (D_MODEL / 16) * 64;

  const int mpair = w & 1;   // phase-1: rows [mpair*32, +32) (2 m-tiles)
  const int nn    = w >> 1;  // phase-1: 16-col tile within 64-col chunk
  const int ncol0 = w * 128; // phase-2: out cols [w*128, +128)

  v8f oacc[4][8];
#pragma unroll
  for (int i = 0; i < 4; ++i)
#pragma unroll
    for (int j = 0; j < 8; ++j) oacc[i][j] = (v8f){};

  for (int hc = 0; hc < D_HIDDEN; hc += 64) {
    // ---- Phase 1: h = gelu(x @ W1[:, hc:hc+64] + b1)
    const int nt1 = (hc >> 4) + nn;                 // global 16-col tile index
    const uint4* w1base = W1fe + ((size_t)nt1 * 32) * 64 + lane * 2;
    if (hc + 64 < D_HIDDEN)
      __builtin_prefetch(W1fe + ((size_t)(nt1 + 4) * 32) * 64 + lane * 2, 0, 1);

    v8f hacc0 = (v8f){}, hacc1 = (v8f){};
    BFrag bf[2];
    bf[0].q[0] = w1base[0];
    bf[0].q[1] = w1base[1];
#pragma unroll
    for (int kt = 0; kt < 32; ++kt) {
      const int cb = kt & 1, nb = cb ^ 1;
      if (kt < 31) {                                // prefetch next fragment
        bf[nb].q[0] = w1base[(kt + 1) * 64 + 0];
        bf[nb].q[1] = w1base[(kt + 1) * 64 + 1];
      }
      AFrag a0, a1;
      a0.q[0] = *(const uint4*)(&XA[(mpair * 2 + 0) * 16 + r][kt * 32 + half * 8]);
      a0.q[1] = *(const uint4*)(&XA[(mpair * 2 + 0) * 16 + r][kt * 32 + 16 + half * 8]);
      a1.q[0] = *(const uint4*)(&XA[(mpair * 2 + 1) * 16 + r][kt * 32 + half * 8]);
      a1.q[1] = *(const uint4*)(&XA[(mpair * 2 + 1) * 16 + r][kt * 32 + 16 + half * 8]);
      hacc0 = __builtin_amdgcn_wmma_f32_16x16x32_bf16(
          false, a0.v, false, bf[cb].v, (short)0, hacc0, false, false);
      hacc1 = __builtin_amdgcn_wmma_f32_16x16x32_bf16(
          false, a1.v, false, bf[cb].v, (short)0, hacc1, false, false);
    }
    {
      const int hcol = hc + nn * 16 + r;
      const float b1c = b1[e * D_HIDDEN + hcol];
#pragma unroll
      for (int v = 0; v < 8; ++v) {
        float h0 = gelu_exact(hacc0[v] + b1c);
        float h1 = gelu_exact(hacc1[v] + b1c);
        Hbuf[(mpair * 2 + 0) * 16 + half * 8 + v][nn * 16 + r] = (__bf16)h0;
        Hbuf[(mpair * 2 + 1) * 16 + half * 8 + v][nn * 16 + r] = (__bf16)h1;
      }
    }
    __syncthreads();

    // ---- Phase 2: oacc += h_chunk @ W2[hc:hc+64, ncol0:ncol0+128]
#pragma unroll
    for (int ks = 0; ks < 2; ++ks) {
      AFrag am[4];
#pragma unroll
      for (int mt = 0; mt < 4; ++mt) {
        am[mt].q[0] = *(const uint4*)(&Hbuf[mt * 16 + r][ks * 32 + half * 8]);
        am[mt].q[1] = *(const uint4*)(&Hbuf[mt * 16 + r][ks * 32 + 16 + half * 8]);
      }
      const int kt2 = (hc >> 5) + ks;               // global 32-row k tile
      const uint4* w2base = W2fe + ((size_t)(w * 8) * 128 + kt2) * 64 + lane * 2;
      BFrag bw[2];
      bw[0].q[0] = w2base[0];
      bw[0].q[1] = w2base[1];
#pragma unroll
      for (int nt = 0; nt < 8; ++nt) {
        const int cb = nt & 1, nb = cb ^ 1;
        if (nt < 7) {                               // prefetch next fragment
          bw[nb].q[0] = w2base[(size_t)(nt + 1) * 128 * 64 + 0];
          bw[nb].q[1] = w2base[(size_t)(nt + 1) * 128 * 64 + 1];
        }
#pragma unroll
        for (int mt = 0; mt < 4; ++mt)
          oacc[mt][nt] = __builtin_amdgcn_wmma_f32_16x16x32_bf16(
              false, am[mt].v, false, bw[cb].v, (short)0, oacc[mt][nt], false,
              false);
      }
    }
    __syncthreads();
  }

  // ---- Epilogue: scale by routing weight, add b2, scatter-add to out
#pragma unroll
  for (int mt = 0; mt < 4; ++mt) {
#pragma unroll
    for (int nt = 0; nt < 8; ++nt) {
      const int ncol = ncol0 + nt * 16 + r;
      const float b2c = b2[e * D_MODEL + ncol];
#pragma unroll
      for (int v = 0; v < 8; ++v) {
        const int i = mt * 16 + half * 8 + v;
        float val = wt_s[i] * (oacc[mt][nt][v] + b2c);
        atomicAdd(out + (size_t)tok_s[i] * D_MODEL + ncol, val);
      }
    }
  }
}

// ---------------------------------------------------------------------------
// Host launcher
// ---------------------------------------------------------------------------
extern "C" void kernel_launch(void* const* d_in, const int* in_sizes, int n_in,
                              void* d_out, int out_size, void* d_ws, size_t ws_size,
                              hipStream_t stream) {
  (void)in_sizes; (void)n_in; (void)out_size; (void)ws_size;

  const float* x  = (const float*)d_in[0];
  const float* Wr = (const float*)d_in[1];
  const float* br = (const float*)d_in[2];
  const float* W1 = (const float*)d_in[3];
  const float* b1 = (const float*)d_in[4];
  const float* W2 = (const float*)d_in[5];
  const float* b2 = (const float*)d_in[6];
  float* out = (float*)d_out;

  // Workspace layout (256B aligned):
  //   [0,256)        : int cnt[8]
  //   +1MB           : int   idx[8][32768]
  //   +1MB           : float wgt[8][32768]
  //   +64MB          : bf16  xb[32768][1024]
  //   +64MB          : bf16  W1f (fragment-swizzled)
  //   +64MB          : bf16  W2f (fragment-swizzled)
  char* ws = (char*)d_ws;
  int*            cnt = (int*)ws;
  int*            idx = (int*)(ws + 256);
  float*          wgt = (float*)(ws + 256 + (1ull << 20));
  unsigned short* xb  = (unsigned short*)(ws + 256 + (2ull << 20));
  unsigned*       w1f = (unsigned*)(ws + 256 + (2ull << 20) + (64ull << 20));
  unsigned*       w2f = (unsigned*)(ws + 256 + (2ull << 20) + (128ull << 20));

  moe_zero_kernel<<<2048, 256, 0, stream>>>(
      (float4*)out, (N_TOKENS * D_MODEL) / 4, cnt);
  moe_cvtx_kernel<<<2048, 256, 0, stream>>>(
      x, (unsigned*)xb, (N_TOKENS * D_MODEL) / 2);
  moe_cvtw_kernel<<<4096, 256, 0, stream>>>(W1, w1f, D_MODEL, D_HIDDEN);
  moe_cvtw_kernel<<<4096, 256, 0, stream>>>(W2, w2f, D_HIDDEN, D_MODEL);
  moe_router_kernel<<<N_TOKENS / 8, 256, 0, stream>>>(x, Wr, br, cnt, idx, wgt);

  dim3 grid(N_TOKENS / 64, N_EXPERTS);
  moe_expert_kernel<<<grid, 256, SM_TOTAL, stream>>>(xb, (const uint4*)w1f, b1,
                                                     (const uint4*)w2f, b2, cnt,
                                                     idx, wgt, out);
}
